// MultiplexGNN_56899726737491
// MI455X (gfx1250) — matrix operand order repaired and checked by reading, not compiled, not run on hardware
//
#include <hip/hip_runtime.h>

typedef __attribute__((ext_vector_type(16))) _Float16 v16h;
typedef __attribute__((ext_vector_type(8)))  float    v8f;
typedef int v4i __attribute__((__vector_size__(16)));

#define NPL     50000
#define NNODES  100000
#define NEDGES  800000
#define F1      64
#define F2      128
#define NEG_SLOPE 0.01f

// workspace layout (float offsets)
#define O_DEGAC 0
#define O_DEGCA (O_DEGAC + NNODES)
#define O_S     (O_DEGCA + NNODES)
#define O_Z1    (O_S + NNODES)                 // 128 slots (100 used)
#define O_H1    (O_Z1 + 128)                   // NNODES*128 accumulator -> h1
#define O_S2AC  (O_H1 + NNODES * F2)
#define O_S2CA  (O_S2AC + NNODES * F2)
#define TOTF    (O_S2CA + NNODES * F2)         // 38,700,128 floats (~154.8 MB)

#if defined(__AMDGCN__) && __has_builtin(__builtin_amdgcn_global_load_async_to_lds_b128) && \
    __has_builtin(__builtin_amdgcn_s_wait_asynccnt)
#define HAVE_ASYNC_LDS 1
// builtin expects pointers to 128-bit int vectors in AS1 (global) / AS3 (LDS)
#define ASG(p) ((__attribute__((address_space(1))) v4i*)(p))
#define ASL(p) ((__attribute__((address_space(3))) v4i*)(p))
#else
#define HAVE_ASYNC_LDS 0
#endif

__device__ __forceinline__ float lrelu(float x) { return x > 0.0f ? x : NEG_SLOPE * x; }

// ---------------- K0: zero workspace ----------------
__global__ void k_zero(float4* ws4, int n4) {
    int i = blockIdx.x * blockDim.x + threadIdx.x;
    if (i < n4) ws4[i] = make_float4(0.f, 0.f, 0.f, 0.f);
}

// ---------------- K1: degrees ----------------
__global__ void k_deg(const int* __restrict__ src, const int* __restrict__ dst,
                      float* degAC, float* degCA) {
    int e = blockIdx.x * blockDim.x + threadIdx.x;
    if (e < NEDGES) {
        atomicAdd(&degAC[dst[e]], 1.0f);
        atomicAdd(&degCA[src[e]], 1.0f);
    }
}

// ---------------- K2: layer-1 edge scatter (one wave per edge) ----------------
__global__ void k_scatter1(const int* __restrict__ src, const int* __restrict__ dst,
                           const float* __restrict__ nf, float* h1acc) {
    int tid  = blockIdx.x * blockDim.x + threadIdx.x;
    int e    = tid >> 5;
    int lane = tid & 31;
    if (e >= NEDGES) return;
    int si = src[e], di = dst[e];
    int srow = si >= NPL ? si - NPL : si;   // feat = tile(node_features, (2,1))
    int drow = di >= NPL ? di - NPL : di;
#pragma unroll
    for (int t = 0; t < 2; ++t) {
        int f = lane + 32 * t;
        atomicAdd(&h1acc[di * F2 + f],      nf[srow * F1 + f]);  // AC half
        atomicAdd(&h1acc[si * F2 + F1 + f], nf[drow * F1 + f]);  // CA half
    }
}

// ---------------- K3: layer-1 finalize in place ----------------
__global__ void k_fin1(const float* __restrict__ nf, const float* __restrict__ degAC,
                       const float* __restrict__ degCA, float* h1) {
    int tid = blockIdx.x * blockDim.x + threadIdx.x;
    int v = tid >> 7, f = tid & 127;
    if (v >= NNODES) return;
    int row = v >= NPL ? v - NPL : v;
    float base = nf[row * F1 + (f & 63)];
    float d = (f < F1) ? degAC[v] : degCA[v];
    d = d < 1.0f ? 1.0f : d;
    int idx = v * F2 + f;
    h1[idx] = lrelu(base + h1[idx] / d);
}

// ---------------- K4: layer-2 edge scatter ----------------
__global__ void k_scatter2(const int* __restrict__ src, const int* __restrict__ dst,
                           const float* __restrict__ h1, float* s2AC, float* s2CA) {
    int tid = blockIdx.x * blockDim.x + threadIdx.x;
    int e = tid >> 7, f = tid & 127;
    if (e >= NEDGES) return;
    int si = src[e], di = dst[e];
    atomicAdd(&s2AC[di * F2 + f], h1[si * F2 + f]);
    atomicAdd(&s2CA[si * F2 + f], h1[di * F2 + f]);
}

// ---------------- K5: per-node scalar s[v] (one wave per node) ----------------
// Rows staged to LDS via gfx1250 async global->LDS b128 when available.
__global__ void k_snode(const float* __restrict__ h1, const float* __restrict__ s2AC,
                        const float* __restrict__ s2CA, const float* __restrict__ degAC,
                        const float* __restrict__ degCA, const float* __restrict__ Wagg,
                        const float* __restrict__ bagg, float* sv) {
    int tid = blockIdx.x * blockDim.x + threadIdx.x;
    int v = tid >> 5, lane = tid & 31;
    if (v >= NNODES) return;
    float dAC = degAC[v]; dAC = dAC < 1.f ? 1.f : dAC; float rAC = 1.0f / dAC;
    float dCA = degCA[v]; dCA = dCA < 1.f ? 1.f : dCA; float rCA = 1.0f / dCA;
    float acc = 0.0f;
    int rowb = v * F2;
#if HAVE_ASYNC_LDS
    __shared__ __align__(16) float lds[8 * 3 * F2];   // 8 waves x (h1,s2AC,s2CA) rows
    float* lH = &lds[(threadIdx.x >> 5) * (3 * F2)];
    float* lA = lH + F2;
    float* lC = lH + 2 * F2;
    // one b128 per lane per row: 32 lanes x 16B = one 128-float row
    __builtin_amdgcn_global_load_async_to_lds_b128(ASG(h1   + rowb + lane * 4),
                                                   ASL(lH + lane * 4), 0, 0);
    __builtin_amdgcn_global_load_async_to_lds_b128(ASG(s2AC + rowb + lane * 4),
                                                   ASL(lA + lane * 4), 0, 0);
    __builtin_amdgcn_global_load_async_to_lds_b128(ASG(s2CA + rowb + lane * 4),
                                                   ASL(lC + lane * 4), 0, 0);
    __builtin_amdgcn_s_wait_asynccnt(0);
#pragma unroll
    for (int t = 0; t < 4; ++t) {
        int f = lane + 32 * t;
        float hv = lH[f];
        acc += lrelu(hv + lA[f] * rAC) * Wagg[f];
        acc += lrelu(hv + lC[f] * rCA) * Wagg[F2 + f];
    }
#else
#pragma unroll
    for (int t = 0; t < 4; ++t) {
        int f = lane + 32 * t;
        int idx = rowb + f;
        float hv = h1[idx];
        acc += lrelu(hv + s2AC[idx] * rAC) * Wagg[f];
        acc += lrelu(hv + s2CA[idx] * rCA) * Wagg[F2 + f];
    }
#endif
#pragma unroll
    for (int off = 16; off > 0; off >>= 1) acc += __shfl_xor(acc, off, 32);
    if (lane == 0) sv[v] = acc + bagg[0];
}

// ---------------- K6: z1acc[j] = sum_v s[v] * W1[v,j] via WMMA ----------------
// A (16x32 f16): one contiguous 16-float run per lane -> float4 loads.
// B (32x16 f16): column gather, branchless (clamped col, guarded commit).
__device__ __forceinline__ v16h mk_a(const float* __restrict__ sv, int k0, int g) {
    const float4* p0 = (const float4*)(sv + k0 + g * 16);
    const float4* p1 = (const float4*)(sv + k0 + g * 16 + 8);
    float4 r0 = p0[0], r1 = p0[1], r2 = p1[0], r3 = p1[1];
    v16h a;
    a[0]  = (_Float16)r0.x; a[1]  = (_Float16)r0.y; a[2]  = (_Float16)r0.z; a[3]  = (_Float16)r0.w;
    a[4]  = (_Float16)r1.x; a[5]  = (_Float16)r1.y; a[6]  = (_Float16)r1.z; a[7]  = (_Float16)r1.w;
    a[8]  = (_Float16)r2.x; a[9]  = (_Float16)r2.y; a[10] = (_Float16)r2.z; a[11] = (_Float16)r2.w;
    a[12] = (_Float16)r3.x; a[13] = (_Float16)r3.y; a[14] = (_Float16)r3.z; a[15] = (_Float16)r3.w;
    return a;
}

__device__ __forceinline__ v16h mk_b(const float* __restrict__ W1, int k0, int g, int colc) {
    int base = (k0 + g * 16) * 100 + colc;
    v16h b;
#pragma unroll
    for (int e = 0; e < 16; ++e) b[e] = (_Float16)W1[base + e * 100];
    return b;
}

__global__ void __launch_bounds__(32) k_matvec_wmma(const float* __restrict__ sv,
                                                    const float* __restrict__ W1,
                                                    float* z1acc) {
    const int lane   = threadIdx.x;            // 0..31
    const int tile   = blockIdx.x;             // 0..6 -> cols [16*tile, 16*tile+15]
    const int k0base = blockIdx.y * 4000;      // node chunk
    const int g      = lane >> 4;              // lane half-group
    const int col    = tile * 16 + (lane & 15);
    const int colc   = col < 100 ? col : 99;   // branchless in-bounds clamp

    v8f c0 = {}, c1 = {};
    for (int it = 0; it < 124; it += 2) {
        int k0 = k0base + it * 32;
        __builtin_prefetch(&W1[(k0 + 64) * 100 + colc], 0, 1);  // global_prefetch_b8
        v16h a0 = mk_a(sv, k0, g);
        v16h b0 = mk_b(W1, k0, g, colc);
        v16h a1 = mk_a(sv, k0 + 32, g);
        v16h b1 = mk_b(W1, k0 + 32, g, colc);
        c0 = __builtin_amdgcn_wmma_f32_16x16x32_f16(false, a0, false, b0, (short)0, c0,
                                                    false, false);
        c1 = __builtin_amdgcn_wmma_f32_16x16x32_f16(false, a1, false, b1, (short)0, c1,
                                                    false, false);
    }
    {   // remainder chunk (125th)
        int k0 = k0base + 124 * 32;
        v16h a0 = mk_a(sv, k0, g);
        v16h b0 = mk_b(W1, k0, g, colc);
        c0 = __builtin_amdgcn_wmma_f32_16x16x32_f16(false, a0, false, b0, (short)0, c0,
                                                    false, false);
    }
    // all C rows identical (A rows replicated); row M=0 lives in c[0] of lanes 0..15
    if (lane < 16 && col < 100) atomicAdd(&z1acc[col], c0[0] + c1[0]);
}

// ---------------- K7: MLP head ----------------
__global__ void k_head(const float* __restrict__ z1acc, const float* __restrict__ b1,
                       const float* __restrict__ W2, const float* __restrict__ b2,
                       const float* __restrict__ W3, const float* __restrict__ b3,
                       float* out) {
    __shared__ float z1[100];
    __shared__ float z2[20];
    int t = threadIdx.x;
    if (t < 100) z1[t] = lrelu(z1acc[t] + b1[t]);
    __syncthreads();
    if (t < 20) {
        float a = b2[t];
        for (int i = 0; i < 100; ++i) a += z1[i] * W2[i * 20 + t];
        z2[t] = lrelu(a);
    }
    __syncthreads();
    if (t < 2) {
        float a = b3[t];
        for (int i = 0; i < 20; ++i) a += z2[i] * W3[i * 2 + t];
        out[t] = a;
    }
}

extern "C" void kernel_launch(void* const* d_in, const int* in_sizes, int n_in,
                              void* d_out, int out_size, void* d_ws, size_t ws_size,
                              hipStream_t stream) {
    const int*   src  = (const int*)d_in[0];
    const int*   dst  = (const int*)d_in[1];
    const float* nf   = (const float*)d_in[2];
    // d_in[3] = M (int scalar) == 2, baked into constants
    const float* Wagg = (const float*)d_in[4];
    const float* bagg = (const float*)d_in[5];
    const float* W1   = (const float*)d_in[6];
    const float* b1   = (const float*)d_in[7];
    const float* W2   = (const float*)d_in[8];
    const float* b2   = (const float*)d_in[9];
    const float* W3   = (const float*)d_in[10];
    const float* b3   = (const float*)d_in[11];

    float* ws    = (float*)d_ws;
    float* degAC = ws + O_DEGAC;
    float* degCA = ws + O_DEGCA;
    float* sv    = ws + O_S;
    float* z1acc = ws + O_Z1;
    float* h1    = ws + O_H1;
    float* s2AC  = ws + O_S2AC;
    float* s2CA  = ws + O_S2CA;

    const int n4 = TOTF / 4;  // TOTF divisible by 4
    k_zero<<<(n4 + 255) / 256, 256, 0, stream>>>((float4*)ws, n4);

    k_deg<<<(NEDGES + 255) / 256, 256, 0, stream>>>(src, dst, degAC, degCA);

    k_scatter1<<<(NEDGES * 32) / 256, 256, 0, stream>>>(src, dst, nf, h1);

    k_fin1<<<(NNODES * F2) / 256, 256, 0, stream>>>(nf, degAC, degCA, h1);

    k_scatter2<<<(NEDGES / 2), 256, 0, stream>>>(src, dst, h1, s2AC, s2CA);  // E*128/256

    k_snode<<<(NNODES * 32) / 256, 256, 0, stream>>>(h1, s2AC, s2CA, degAC, degCA,
                                                     Wagg, bagg, sv);

    dim3 gmv(7, 25);
    k_matvec_wmma<<<gmv, 32, 0, stream>>>(sv, W1, z1acc);

    k_head<<<1, 128, 0, stream>>>(z1acc, b1, W2, b2, W3, b3, (float*)d_out);
}
